// Generator_16389595202101
// MI455X (gfx1250) — matrix operand order repaired and checked by reading, not compiled, not run on hardware
//
#include <hip/hip_runtime.h>

// MI455X / gfx1250: wave32, WMMA 16x16x32 f16->f32, 320KB LDS per WGP, 192MB L2.
// Persistent-block GRU scan: 8 blocks x 16 batch rows, 256 threads (8 waves),
// all T=128 steps inside one kernel; GEMMs via v_wmma_f32_16x16x32_f16.
// Weights pre-converted to f16 in d_ws (halves streamed bytes, removes cvt
// chain in front of the WMMAs); canvas stores are non-temporal so the 32MB
// output stream does not evict L2-resident weights.

typedef __attribute__((ext_vector_type(16))) _Float16 v16h;
typedef __attribute__((ext_vector_type(8)))  float    v8f;

#define NTHREADS 256
#define NWAVES   8
#define MROWS    16      // batch rows per block (== WMMA M)
#define N_H      512
#define T_STEPS  128
#define F_FRAMES 65536
#define V_EMB    4096

#define W1_ELEMS (512 * 1024)
#define WI_ELEMS (1536 * 512)
#define WH_ELEMS (1536 * 512)
#define WS_HALF_ELEMS (W1_ELEMS + WI_ELEMS + WH_ELEMS)   // 2,097,152 -> 4MB f16

// ---- dynamic LDS layout ----
#define SMEM_BYTES (98304 + 4*32768 + 1024 + 1024 + 64 + 64 + 32768 + 256)

__device__ __forceinline__ float sigf(float x) { return 1.0f / (1.0f + expf(-x)); }

// One 16xN GEMM: D[m][ncol] = sum_k A[m][aoff+k] * W[ncol][k] + bias[ncol]
// A: f16 row-major in LDS. W: row-major global, f16 (fast path) or f32.
// N-tiles round-robined across the 8 waves; K in steps of 32.
// Columns >= split go to out2 (the GRU "hn" slice of gh).
template <typename WT>
__device__ __forceinline__ void gemm16(
    const _Float16* __restrict__ sA, int aoff, int lda, int Ktot,
    const WT* __restrict__ W, int ldw,
    const float* __restrict__ bias,
    int Ncols, int split,
    float* __restrict__ out, int ldo, bool accum,
    float* __restrict__ out2, int ld2,
    int lane, int wave)
{
  const int m      = lane & 15;   // A-matrix row owned by this lane
  const int grp    = lane >> 4;   // lane-half selects K-subgroup
  const int nloc   = lane & 15;   // B/C/D column within tile
  const int ntiles = Ncols >> 4;
  const int ksteps = Ktot >> 5;

  for (int nt = wave; nt < ntiles; nt += NWAVES) {
    const int ncol = nt * 16 + nloc;
    const WT* wrow = W + (size_t)ncol * ldw;
    v8f c = {0.f, 0.f, 0.f, 0.f, 0.f, 0.f, 0.f, 0.f};

    for (int ks = 0; ks < ksteps; ++ks) {
      const int kb = ks << 5;
      v16h a, b;
      // A 16x32 f16 layout (matches B): lanes 0-15 K=0..15,
      // lanes 16-31 K=16..31 -> base offset grp*16, contiguous.
      const _Float16* p = sA + m * lda + aoff + kb + grp * 16;
      a[0]=p[0];  a[1]=p[1];  a[2]=p[2];  a[3]=p[3];
      a[4]=p[4];  a[5]=p[5];  a[6]=p[6];  a[7]=p[7];
      a[8]=p[8]; a[9]=p[9]; a[10]=p[10]; a[11]=p[11];
      a[12]=p[12];a[13]=p[13];a[14]=p[14]; a[15]=p[15];
      // B 32x16: lanes 0-15 hold K=0..15 of column nloc, lanes 16-31 K=16..31.
      if constexpr (sizeof(WT) == 2) {
        // f16 weights: one aligned 32B fragment load, nothing between it
        // and the WMMA.
        b = *(const v16h*)((const _Float16*)wrow + kb + grp * 16);
      } else {
        const float4* q = (const float4*)((const float*)wrow + kb + grp * 16);
        float4 q0=q[0], q1=q[1], q2=q[2], q3=q[3];
        b[0]=(_Float16)q0.x; b[1]=(_Float16)q0.y; b[2]=(_Float16)q0.z; b[3]=(_Float16)q0.w;
        b[4]=(_Float16)q1.x; b[5]=(_Float16)q1.y; b[6]=(_Float16)q1.z; b[7]=(_Float16)q1.w;
        b[8]=(_Float16)q2.x; b[9]=(_Float16)q2.y; b[10]=(_Float16)q2.z; b[11]=(_Float16)q2.w;
        b[12]=(_Float16)q3.x; b[13]=(_Float16)q3.y; b[14]=(_Float16)q3.z; b[15]=(_Float16)q3.w;
      }

      c = __builtin_amdgcn_wmma_f32_16x16x32_f16(
            /*neg_a=*/false, a, /*neg_b=*/false, b,
            /*c_mod=*/(short)0, c, /*reuse_a=*/false, /*reuse_b=*/false);
    }
    // D f32 16x16: VGPR r -> row r (lanes 0-15) / row r+8 (lanes 16-31)
#pragma unroll
    for (int r = 0; r < 8; ++r) {
      const int mo = r + grp * 8;
      const float v = c[r] + bias[ncol];
      if (ncol < split) {
        if (accum) out[mo * ldo + ncol] += v;
        else       out[mo * ldo + ncol]  = v;
      } else {
        out2[mo * ld2 + (ncol - split)] = v;
      }
    }
  }
}

// Convert the three GEMM weight matrices to f16 into workspace (once per call).
__global__ void __launch_bounds__(NTHREADS)
cvt_weights_kernel(const float* __restrict__ w_mt1,
                   const float* __restrict__ w_ih,
                   const float* __restrict__ w_hh,
                   _Float16* __restrict__ ws)
{
  const int total = WS_HALF_ELEMS;
  for (int e = blockIdx.x * blockDim.x + threadIdx.x; e < total;
       e += gridDim.x * blockDim.x) {
    float v;
    if (e < W1_ELEMS)                 v = w_mt1[e];
    else if (e < W1_ELEMS + WI_ELEMS) v = w_ih[e - W1_ELEMS];
    else                              v = w_hh[e - W1_ELEMS - WI_ELEMS];
    ws[e] = (_Float16)v;
  }
}

template <typename WT>
__global__ void __launch_bounds__(NTHREADS)
gru_scan_kernel(const float* __restrict__ z,        // (128,128,511)
                const int*   __restrict__ alpha,    // (128,128)
                const float* __restrict__ conv_w,   // (5)
                const float* __restrict__ conv_b,   // (1)
                const WT*    __restrict__ w_mt1,    // (512,1024)
                const float* __restrict__ b_mt1,    // (512)
                const WT*    __restrict__ w_ih,     // (1536,512)
                const WT*    __restrict__ w_hh,     // (1536,512)
                const float* __restrict__ b_ih,     // (1536)
                const float* __restrict__ b_hh,     // (1536)
                const float* __restrict__ w_ht,     // (1,512)
                const float* __restrict__ b_ht,     // (1)
                const float* __restrict__ emb,      // (4096,512)
                const float* __restrict__ w_ct,     // (1,513)
                const float* __restrict__ b_ct,     // (1)
                float*       __restrict__ out)      // (128,65536)
{
  extern __shared__ char smem[];
  float*    gacc = (float*)smem;                          // 16x1536
  float*    st1  = gacc + MROWS * 1536;                   // 16x512
  float*    ht1  = st1  + MROWS * N_H;                    // 16x512
  float*    Hst  = ht1  + MROWS * N_H;                    // 16x512
  float*    hn   = Hst  + MROWS * N_H;                    // 16x512
  float*    red1 = hn   + MROWS * N_H;                    // 256
  float*    red2 = red1 + NTHREADS;                       // 256
  float*    dafs = red2 + NTHREADS;                       // 16
  float*    ctv  = dafs + MROWS;                          // 16
  _Float16* sA   = (_Float16*)(ctv + MROWS);              // 16x1024 f16
  int*      itb   = (int*)(sA + MROWS * 1024);            // 16
  int*      a_cur = itb + MROWS;                          // 16
  int*      a_nxt = a_cur + MROWS;                        // 16
  int*      da_i  = a_nxt + MROWS;                        // 16

  const int tid   = threadIdx.x;
  const int lane  = tid & 31;
  const int wave  = tid >> 5;
  const int bBase = blockIdx.x * MROWS;

  const float cw0 = conv_w[0], cw1 = conv_w[1], cw2 = conv_w[2],
              cw3 = conv_w[3], cw4 = conv_w[4], cb = conv_b[0];

  // carries start at zero
  for (int e = tid; e < MROWS * N_H; e += NTHREADS) { st1[e] = 0.f; ht1[e] = 0.f; }
  __syncthreads();

  for (int t = 0; t < T_STEPS; ++t) {
    // -- phase 0: alpha / da per row --------------------------------------
    if (tid < MROWS) {
      const int b  = bBase + tid;
      const int at = alpha[b * T_STEPS + t];
      const int ap = (t == 0) ? 0 : alpha[b * T_STEPS + t - 1];
      a_cur[tid] = at;
      a_nxt[tid] = (t == T_STEPS - 1) ? F_FRAMES : alpha[b * T_STEPS + t + 1];
      da_i[tid]  = at - ap;
      dafs[tid]  = (float)(at - ap);
    }
    __syncthreads();

    // -- phase 1: Mt1 = [conv1d(st1), ht1]; sA = relu(Mt1) as f16 ----------
    for (int e = tid; e < MROWS * N_H; e += NTHREADS) {
      const int r = e >> 9, j = e & (N_H - 1);
      const float* s = st1 + r * N_H;
      float c = cb;
      if (j >= 2)       c += cw0 * s[j - 2];
      if (j >= 1)       c += cw1 * s[j - 1];
      c += cw2 * s[j];
      if (j + 1 < N_H)  c += cw3 * s[j + 1];
      if (j + 2 < N_H)  c += cw4 * s[j + 2];
      sA[r * 1024 + j]       = (_Float16)fmaxf(c, 0.f);
      sA[r * 1024 + 512 + j] = (_Float16)fmaxf(ht1[e], 0.f);
    }
    __syncthreads();

    // -- phase 2: Hst = relu(Mt1) @ w_mt1.T + b_mt1  (K=1024, N=512) -------
    gemm16(sA, 0, 1024, 1024, w_mt1, 1024, b_mt1, 512, 512,
           Hst, N_H, false, hn, N_H, lane, wave);
    __syncthreads();

    // -- phase 3: sA = [xt | Hst] as f16 -----------------------------------
    for (int e = tid; e < MROWS * N_H; e += NTHREADS) {
      const int r = e >> 9, j = e & (N_H - 1);
      const int b = bBase + r;
      const float xv = (j < N_H - 1)
          ? z[((size_t)b * T_STEPS + t) * (N_H - 1) + j]
          : dafs[r];
      sA[r * 1024 + j]       = (_Float16)xv;
      sA[r * 1024 + 512 + j] = (_Float16)Hst[e];
    }
    __syncthreads();

    // -- phase 4: gacc = gi = xt @ w_ih.T + b_ih  (K=512, N=1536) ----------
    gemm16(sA, 0, 1024, 512, w_ih, 512, b_ih, 1536, 1536,
           gacc, 1536, false, hn, N_H, lane, wave);
    __syncthreads();

    // -- phase 5: gh = Hst @ w_hh.T + b_hh; gacc += gh (cols<1024), hn rest
    gemm16(sA, 512, 1024, 512, w_hh, 512, b_hh, 1536, 1024,
           gacc, 1536, true, hn, N_H, lane, wave);
    __syncthreads();

    // -- phase 6: GRU gates -> ht1 ----------------------------------------
    for (int e = tid; e < MROWS * N_H; e += NTHREADS) {
      const int r = e >> 9, j = e & (N_H - 1);
      const float rg = sigf(gacc[r * 1536 + j]);
      const float zg = sigf(gacc[r * 1536 + 512 + j]);
      const float ng = tanhf(gacc[r * 1536 + 1024 + j] + rg * hn[e]);
      ht1[e] = (1.f - zg) * ng + zg * Hst[e];
    }
    __syncthreads();

    // -- phase 7: row dots -> it (emb index), ct ---------------------------
    {
      const int r = tid >> 4, l16 = tid & 15;
      float s1 = 0.f, s2 = 0.f;
      for (int j = l16; j < N_H; j += 16) {
        const float hv = ht1[r * N_H + j];
        s1 += hv * w_ht[j];
        s2 += fmaxf(hv, 0.f) * w_ct[j];
      }
      red1[tid] = s1; red2[tid] = s2;
    }
    __syncthreads();
    if (tid < MROWS) {
      float s1 = 0.f, s2 = 0.f;
#pragma unroll
      for (int i = 0; i < 16; ++i) { s1 += red1[tid * 16 + i]; s2 += red2[tid * 16 + i]; }
      const float logit = s1 + b_ht[0];
      itb[tid] = (int)((float)(V_EMB - 1) * sigf(logit));
      ctv[tid] = sigf(s2 + fmaxf(dafs[tid], 0.f) * w_ct[N_H] + b_ct[0]);
    }
    __syncthreads();

    // -- phase 8: st = shift(st1, da)*ct + emb[it]  (regs, then writeback) -
    {
      float regs[(MROWS * N_H) / NTHREADS];
#pragma unroll
      for (int i = 0; i < (MROWS * N_H) / NTHREADS; ++i) {
        const int e = tid + i * NTHREADS;
        const int r = e >> 9, j = e & (N_H - 1);
        const int idx = j + da_i[r];
        const float frag = (idx < N_H) ? st1[r * N_H + idx] : 0.f;
        regs[i] = frag * ctv[r] + emb[(size_t)itb[r] * N_H + j];
      }
      __syncthreads();
#pragma unroll
      for (int i = 0; i < (MROWS * N_H) / NTHREADS; ++i)
        st1[tid + i * NTHREADS] = regs[i];
    }
    __syncthreads();

    // -- phase 9: canvas scatter; region [a_t, a_{t+1}) is final at step t.
    // Non-temporal stores: output is write-once, keep L2 for the weights.
    for (int r = 0; r < MROWS; ++r) {
      const int b = bBase + r;
      const int a0 = a_cur[r], stop = a_nxt[r];
      if (t == 0)  // prefix [0, a_0) stays zero forever
        for (int f = tid; f < a0; f += NTHREADS)
          __builtin_nontemporal_store(0.f, &out[(size_t)b * F_FRAMES + f]);
      for (int f = a0 + tid; f < stop; f += NTHREADS) {
        const int rel = f - a0;
        const float v = (rel < N_H) ? st1[r * N_H + rel] : 0.f;
        __builtin_nontemporal_store(v, &out[(size_t)b * F_FRAMES + f]);
      }
    }
    __syncthreads();
  }
}

extern "C" void kernel_launch(void* const* d_in, const int* in_sizes, int n_in,
                              void* d_out, int out_size, void* d_ws, size_t ws_size,
                              hipStream_t stream) {
  (void)in_sizes; (void)n_in; (void)out_size;
  const float* zz     = (const float*)d_in[0];
  const int*   alpha  = (const int*)  d_in[1];
  const float* conv_w = (const float*)d_in[2];
  const float* conv_b = (const float*)d_in[3];
  const float* w_mt1  = (const float*)d_in[4];
  const float* b_mt1  = (const float*)d_in[5];
  const float* w_ih   = (const float*)d_in[6];
  const float* w_hh   = (const float*)d_in[7];
  const float* b_ih   = (const float*)d_in[8];
  const float* b_hh   = (const float*)d_in[9];
  const float* w_ht   = (const float*)d_in[10];
  const float* b_ht   = (const float*)d_in[11];
  const float* emb    = (const float*)d_in[12];
  const float* w_ct   = (const float*)d_in[13];
  const float* b_ct   = (const float*)d_in[14];
  float* o = (float*)d_out;

  const size_t need = (size_t)WS_HALF_ELEMS * sizeof(_Float16);
  if (ws_size >= need && d_ws != nullptr) {
    _Float16* wsH = (_Float16*)d_ws;
    cvt_weights_kernel<<<dim3(512), dim3(NTHREADS), 0, stream>>>(
        w_mt1, w_ih, w_hh, wsH);
    gru_scan_kernel<_Float16>
        <<<dim3(128 / MROWS), dim3(NTHREADS), SMEM_BYTES, stream>>>(
            zz, alpha, conv_w, conv_b,
            wsH, b_mt1,
            wsH + W1_ELEMS, wsH + W1_ELEMS + WI_ELEMS,
            b_ih, b_hh, w_ht, b_ht, emb, w_ct, b_ct, o);
  } else {
    gru_scan_kernel<float>
        <<<dim3(128 / MROWS), dim3(NTHREADS), SMEM_BYTES, stream>>>(
            zz, alpha, conv_w, conv_b,
            w_mt1, b_mt1, w_ih, w_hh,
            b_ih, b_hh, w_ht, b_ht, emb, w_ct, b_ct, o);
  }
}
